// ModelVAE_7078106103848
// MI455X (gfx1250) — matrix operand order repaired and checked by reading, not compiled, not run on hardware
//
#include <hip/hip_runtime.h>
#include <hip/hip_bf16.h>

// ---------------------------------------------------------------------------
// GCN-VAE encoder for MI455X (gfx1250, wave32, WMMA).
//   z_mean, z_var, z = VGAE-encoder(x, edge_index, W1, b1, W_mu, b_mu, W_var, b_var, eps)
// GEMMs use v_wmma_f32_16x16x32_bf16 (bf16 in / f32 accumulate); graph
// aggregation uses f32 global atomics (feature matrices are L2-resident:
// 25.6 MB << 192 MB L2).
// ---------------------------------------------------------------------------

typedef __attribute__((ext_vector_type(16))) __bf16 v16bf;
typedef __attribute__((ext_vector_type(8)))  float  v8f;

union BFrag {
    v16bf  v;
    uint4  q[2];     // 32 bytes
    __bf16 e[16];
};

__device__ inline void cvt4(BFrag& f, int o, float4 a) {
    f.e[o + 0] = (__bf16)a.x;
    f.e[o + 1] = (__bf16)a.y;
    f.e[o + 2] = (__bf16)a.z;
    f.e[o + 3] = (__bf16)a.w;
}

// ---------------------------------------------------------------------------
// C[M x 64] = A[M x K] @ W[K x 64], A/W fp32 row-major, bf16 WMMA, f32 accum.
// 256 threads = 8 waves; each wave: one 16-row tile x 64 cols (4 v8f accums).
// W staged in LDS pre-swizzled into B-fragment order: fragment (ks, nt) for
// k-step ks (32 wide) and n-tile nt (16 wide); lane l holds column
// n = nt*16 + (l&15), K = ks*32 + (l>>4)*16 + j for j = 0..15 (32B contiguous).
// ---------------------------------------------------------------------------
__global__ void gemm_bf16_kernel(const float* __restrict__ A,
                                 const float* __restrict__ W,
                                 float* __restrict__ C, int M, int K) {
    extern __shared__ char smem[];        // (K/32)*4 fragments * 1 KB
    const int t     = threadIdx.x;
    const int nk    = K >> 5;             // k-steps of 32
    const int nfrag = nk * 4;

    // ---- stage W -> LDS in WMMA B-fragment layout (bf16) ----
    for (int c = t; c < nfrag * 32; c += 256) {
        int frag  = c >> 5;
        int lane  = c & 31;
        int ks    = frag >> 2;
        int nt    = frag & 3;
        int n     = nt * 16 + (lane & 15);
        int kbase = ks * 32 + (lane >> 4) * 16;
        BFrag f;
#pragma unroll
        for (int j = 0; j < 16; ++j)
            f.e[j] = (__bf16)W[(size_t)(kbase + j) * 64 + n];
        uint4* dq = (uint4*)(smem + (size_t)frag * 1024 + lane * 32);
        dq[0] = f.q[0];
        dq[1] = f.q[1];
    }
    __syncthreads();

    const int wave = t >> 5;
    const int lane = t & 31;
    const int row0 = (blockIdx.x * 8 + wave) * 16;
    if (row0 >= M) return;               // wave-uniform skip (after barrier)

    const int mrow = row0 + (lane & 15);
    const int r    = (mrow < M) ? mrow : (M - 1);  // clamp: EXEC stays all-1
    const int kg   = lane >> 4;

    v8f acc0 = {}, acc1 = {}, acc2 = {}, acc3 = {};

    for (int ks = 0; ks < nk; ++ks) {
        // A fragment: lane-group kg holds K = k0+kg*16 .. +15
        const int k0 = ks * 32;
        const float* ap = A + (size_t)r * K + k0 + kg * 16;
        float4 a0 = *(const float4*)(ap + 0);
        float4 a1 = *(const float4*)(ap + 4);
        float4 a2 = *(const float4*)(ap + 8);
        float4 a3 = *(const float4*)(ap + 12);
        BFrag af;
        cvt4(af, 0, a0); cvt4(af, 4, a1); cvt4(af, 8, a2); cvt4(af, 12, a3);
        v16bf a = af.v;

        const char* bb = smem + (size_t)(ks * 4) * 1024 + lane * 32;
        v16bf b0 = *(const v16bf*)(bb);
        v16bf b1 = *(const v16bf*)(bb + 1024);
        v16bf b2 = *(const v16bf*)(bb + 2048);
        v16bf b3 = *(const v16bf*)(bb + 3072);

        acc0 = __builtin_amdgcn_wmma_f32_16x16x32_bf16(false, a, false, b0,
                                                       (short)0, acc0, false, false);
        acc1 = __builtin_amdgcn_wmma_f32_16x16x32_bf16(false, a, false, b1,
                                                       (short)0, acc1, false, false);
        acc2 = __builtin_amdgcn_wmma_f32_16x16x32_bf16(false, a, false, b2,
                                                       (short)0, acc2, false, false);
        acc3 = __builtin_amdgcn_wmma_f32_16x16x32_bf16(false, a, false, b3,
                                                       (short)0, acc3, false, false);
    }

    // C/D layout: VGPR v -> row = row0 + kg*8 + v, col = (lane&15) + nt*16
    const int col = lane & 15;
#pragma unroll
    for (int v = 0; v < 8; ++v) {
        int row = row0 + kg * 8 + v;
        if (row < M) {
            float* cp = C + (size_t)row * 64 + col;
            cp[0]  = acc0[v];
            cp[16] = acc1[v];
            cp[32] = acc2[v];
            cp[48] = acc3[v];
        }
    }
}

// ---- degree: deg[dst] += 1 per edge (edge_index is int64 per reference) ----
__global__ void degree_kernel(const long long* __restrict__ dst,
                              float* __restrict__ deg, int E) {
    int i = blockIdx.x * 256 + threadIdx.x;
    if (i < E) atomicAdd(&deg[(int)dst[i]], 1.0f);
}

// ---- deg -> dinv = rsqrt(deg + 1) in place ----
__global__ void dinv_kernel(float* __restrict__ deg, int N) {
    int i = blockIdx.x * 256 + threadIdx.x;
    if (i < N) deg[i] = rsqrtf(deg[i] + 1.0f);
}

// ---- scatter: agg[dst,:] += xw[src,:] * dinv[src]*dinv[dst]; 32 lanes/edge,
//      each lane owns 2 features (float2, coalesced 256B per edge) ----
__global__ void scatter_kernel(const long long* __restrict__ src,
                               const long long* __restrict__ dst,
                               const float* __restrict__ xw,
                               const float* __restrict__ dinv,
                               float* __restrict__ agg, int E) {
    long long tid = (long long)blockIdx.x * 256 + threadIdx.x;
    int e = (int)(tid >> 5);
    if (e >= E) return;
    int f = ((int)tid & 31) * 2;
    int s = (int)src[e], d = (int)dst[e];
    float nrm = dinv[s] * dinv[d];
    float2 v = *(const float2*)(xw + (size_t)s * 64 + f);
    float* op = agg + (size_t)d * 64 + f;
    atomicAdd(op + 0, v.x * nrm);
    atomicAdd(op + 1, v.y * nrm);
}

// ---- dual scatter for mu/var into d_out regions ----
__global__ void scatter2_kernel(const long long* __restrict__ src,
                                const long long* __restrict__ dst,
                                const float* __restrict__ am,
                                const float* __restrict__ av,
                                const float* __restrict__ dinv,
                                float* __restrict__ om,
                                float* __restrict__ ov, int E) {
    long long tid = (long long)blockIdx.x * 256 + threadIdx.x;
    int e = (int)(tid >> 5);
    if (e >= E) return;
    int f = ((int)tid & 31) * 2;
    int s = (int)src[e], d = (int)dst[e];
    float nrm = dinv[s] * dinv[d];
    float2 vm = *(const float2*)(am + (size_t)s * 64 + f);
    float2 vv = *(const float2*)(av + (size_t)s * 64 + f);
    size_t off = (size_t)d * 64 + f;
    atomicAdd(om + off + 0, vm.x * nrm);
    atomicAdd(om + off + 1, vm.y * nrm);
    atomicAdd(ov + off + 0, vv.x * nrm);
    atomicAdd(ov + off + 1, vv.y * nrm);
}

// ---- h = relu(agg + xw * dinv^2 + b), in place over agg ----
__global__ void combine_relu_kernel(float* __restrict__ agg,
                                    const float* __restrict__ xw,
                                    const float* __restrict__ dinv,
                                    const float* __restrict__ b, int N) {
    long long i = (long long)blockIdx.x * 256 + threadIdx.x;
    if (i >= (long long)N * 64) return;
    int row = (int)(i >> 6), c = (int)(i & 63);
    float dv = dinv[row];
    float v  = agg[i] + xw[i] * dv * dv + b[c];
    agg[i] = v > 0.0f ? v : 0.0f;
}

// ---- finalize: self-loop + bias, softplus, reparameterize ----
__global__ void finalize_kernel(float* __restrict__ zm, float* __restrict__ zv,
                                float* __restrict__ z,
                                const float* __restrict__ hwmu,
                                const float* __restrict__ hwvar,
                                const float* __restrict__ dinv,
                                const float* __restrict__ bmu,
                                const float* __restrict__ bvar,
                                const float* __restrict__ eps, int N) {
    long long i = (long long)blockIdx.x * 256 + threadIdx.x;
    if (i >= (long long)N * 64) return;
    int row = (int)(i >> 6), c = (int)(i & 63);
    float dv = dinv[row], d2 = dv * dv;
    float m  = zm[i] + hwmu[i] * d2 + bmu[c];
    float vp = zv[i] + hwvar[i] * d2 + bvar[c];
    float sv = (vp > 20.0f) ? vp : log1pf(expf(vp));   // softplus
    zm[i] = m;
    zv[i] = sv;
    z[i]  = m + sv * eps[i];
}

extern "C" void kernel_launch(void* const* d_in, const int* in_sizes, int n_in,
                              void* d_out, int out_size, void* d_ws, size_t ws_size,
                              hipStream_t stream) {
    const float*     x    = (const float*)d_in[0];
    const long long* ei   = (const long long*)d_in[1];   // int64 per reference
    const float*     W1   = (const float*)d_in[2];
    const float*     b1   = (const float*)d_in[3];
    const float*     Wmu  = (const float*)d_in[4];
    const float*     bmu  = (const float*)d_in[5];
    const float*     Wvar = (const float*)d_in[6];
    const float*     bvar = (const float*)d_in[7];
    const float*     eps  = (const float*)d_in[8];

    const int IDIM = 512, H = 64;
    const int N = in_sizes[0] / IDIM;
    const int E = in_sizes[1] / 2;
    const long long NH = (long long)N * H;

    const long long* srcp = ei;
    const long long* dstp = ei + E;

    // workspace: dinv [N] | xw1 [N*64] | h/agg [N*64] | hwmu [N*64] | hwvar [N*64]
    float* ws    = (float*)d_ws;
    float* dinv  = ws;
    float* xw1   = ws + (((size_t)N + 63) & ~(size_t)63);
    float* hbuf  = xw1 + NH;
    float* hwmu  = hbuf + NH;
    float* hwvar = hwmu + NH;

    float* out   = (float*)d_out;
    float* zmean = out;
    float* zvar  = out + NH;
    float* z     = out + 2 * NH;

    hipMemsetAsync(dinv,  0, (size_t)N * 4, stream);
    hipMemsetAsync(hbuf,  0, (size_t)NH * 4, stream);
    hipMemsetAsync(zmean, 0, (size_t)2 * NH * 4, stream);   // zmean + zvar agg

    degree_kernel<<<(E + 255) / 256, 256, 0, stream>>>(dstp, dinv, E);
    dinv_kernel<<<(N + 255) / 256, 256, 0, stream>>>(dinv, N);

    const int gblocks = (N + 127) / 128;        // 8 waves * 16 rows per block
    const long long sthreads = (long long)E * 32;
    const int sblocks = (int)((sthreads + 255) / 256);
    const int nhblocks = (int)((NH + 255) / 256);

    // layer 1: xw1 = x @ W1   (LDS: 512*128 = 64 KB of B-fragments)
    gemm_bf16_kernel<<<gblocks, 256, (size_t)IDIM * 128, stream>>>(x, W1, xw1, N, IDIM);
    scatter_kernel<<<sblocks, 256, 0, stream>>>(srcp, dstp, xw1, dinv, hbuf, E);
    combine_relu_kernel<<<nhblocks, 256, 0, stream>>>(hbuf, xw1, dinv, b1, N);

    // layer 2: mu / var heads  (LDS: 64*128 = 8 KB each)
    gemm_bf16_kernel<<<gblocks, 256, (size_t)H * 128, stream>>>(hbuf, Wmu,  hwmu,  N, H);
    gemm_bf16_kernel<<<gblocks, 256, (size_t)H * 128, stream>>>(hbuf, Wvar, hwvar, N, H);
    scatter2_kernel<<<sblocks, 256, 0, stream>>>(srcp, dstp, hwmu, hwvar, dinv,
                                                 zmean, zvar, E);
    finalize_kernel<<<nhblocks, 256, 0, stream>>>(zmean, zvar, z, hwmu, hwvar,
                                                  dinv, bmu, bvar, eps, N);
}
